// RPEMultiHeadAttention_81123342286997
// MI455X (gfx1250) — matrix operand order repaired and checked by reading, not compiled, not run on hardware
//
#include <hip/hip_runtime.h>
#include <hip/hip_bf16.h>
#include <stdint.h>

typedef __attribute__((ext_vector_type(16))) _Float16     v16h;
typedef __attribute__((ext_vector_type(8)))  float        v8f;
typedef __attribute__((ext_vector_type(4)))  unsigned int u32x4;
typedef __attribute__((ext_vector_type(4)))  float        f32x4;
typedef _Float16 half_t;

#define D_MODEL 64
#define N_HEAD  8
#define BATCH   8
#define SEQ     1024
#define ROWS    (BATCH * SEQ)          // 8192
#define DALL    (D_MODEL * N_HEAD)     // 512
#define INVSCALE 0.07216878364870322f  // 1/sqrt(3*64)

// ---------------------------------------------------------------------------
// WMMA fragment loaders (CDNA5 16-bit layouts, wave32). Key fact: both A and B
// per-lane data are two contiguous 16-byte chunks -> b128 loads.
// A (16x32): lane L -> row m=L%16; chunk0 = K[8hf .. 8hf+7], chunk1 = K[8hf+16 .. +23]
// B (32x16): lane L -> col n=L%16; same K chunks as A: K[8hf .. 8hf+7], K[8hf+16 .. +23]
// C (16x16): lane L -> col n=L%16; vgpr r -> row M = r + 8*hf
// ---------------------------------------------------------------------------
__device__ __forceinline__ v16h load_a16_vec(const half_t* base, int ld, int lane, int kb) {
    int hf = lane >> 4, m = lane & 15;
    const half_t* rp = base + (size_t)m * ld + kb + 8 * hf;
    union { v16h v; u32x4 q[2]; } f;
    f.q[0] = *(const u32x4*)(rp);        // K +0..7
    f.q[1] = *(const u32x4*)(rp + 16);   // K +16..23
    return f.v;
}

__device__ __forceinline__ v16h load_b16_vec(const half_t* base, int ld, int lane, int kb) {
    int hf = lane >> 4, n = lane & 15;
    const half_t* rp = base + (size_t)n * ld + kb + 8 * hf;
    union { v16h v; u32x4 q[2]; } f;
    f.q[0] = *(const u32x4*)(rp);
    f.q[1] = *(const u32x4*)(rp + 16);
    return f.v;
}

__device__ __forceinline__ v16h load_a_f32v(const float* base, int ld, int lane, int kb) {
    int hf = lane >> 4, m = lane & 15;
    const float* rp = base + (size_t)m * ld + kb + 8 * hf;
    f32x4 x0 = *(const f32x4*)(rp);
    f32x4 x1 = *(const f32x4*)(rp + 4);
    f32x4 y0 = *(const f32x4*)(rp + 16);
    f32x4 y1 = *(const f32x4*)(rp + 20);
    union { v16h v; half_t h[16]; } f;
#pragma unroll
    for (int i = 0; i < 4; ++i) {
        f.h[i]      = (half_t)x0[i];
        f.h[4 + i]  = (half_t)x1[i];
        f.h[8 + i]  = (half_t)y0[i];
        f.h[12 + i] = (half_t)y1[i];
    }
    return f.v;
}

__device__ __forceinline__ v8f wmma_f16(v16h a, v16h b, v8f c) {
    return __builtin_amdgcn_wmma_f32_16x16x32_f16(false, a, false, b, (short)0, c, false, false);
}

__device__ __forceinline__ float hmax16(float x) {
#pragma unroll
    for (int m = 1; m < 16; m <<= 1) x = fmaxf(x, __shfl_xor(x, m, 32));
    return x;
}
__device__ __forceinline__ float hsum16(float x) {
#pragma unroll
    for (int m = 1; m < 16; m <<= 1) x += __shfl_xor(x, m, 32);
    return x;
}

__device__ __forceinline__ uint32_t lds_off(const void* p) {
    // flat aperture: low 32 bits of a generic LDS address are the LDS byte address
    return (uint32_t)(uintptr_t)p;
}

// ---------------------------------------------------------------------------
// Kernel 0a: k_r / q_r degenerate-gather vectors (64 threads)
// ---------------------------------------------------------------------------
__global__ void rpe_relvec_kernel(const float* __restrict__ p,
                                  const float* __restrict__ Wkr, const float* __restrict__ bkr,
                                  const float* __restrict__ Wqr, const float* __restrict__ bqr,
                                  float* __restrict__ kr32, float* __restrict__ qr32) {
    int j = threadIdx.x;  // 0..63
    float a = bkr[j], b = bqr[j];
#pragma unroll 8
    for (int i = 0; i < 64; ++i) {
        float pi = p[32 * 64 + i];
        a = fmaf(pi, Wkr[i * 64 + j], a);
        b = fmaf(pi, Wqr[i * 64 + j], b);
    }
    kr32[j] = a;
    qr32[j] = b;
}

// ---------------------------------------------------------------------------
// Kernel 0b: weight pre-transpose to f16 column-major (B-fragment friendly)
// Wt:  [3][512 cols][64 rows]   from Wq/Wk/Wv [64][512]
// Wot: [64 cols][512 rows]      from Wo [512][64]
// ---------------------------------------------------------------------------
__global__ void rpe_wtrans_kernel(const float* __restrict__ Wq, const float* __restrict__ Wk,
                                  const float* __restrict__ Wv, const float* __restrict__ Wo,
                                  half_t* __restrict__ Wt, half_t* __restrict__ Wot) {
    int idx = blockIdx.x * 256 + threadIdx.x;  // 0 .. 131071
    if (idx < 3 * DALL * D_MODEL) {            // 98304: QKV
        int row = idx & 63;                    // k (0..63)
        int col = (idx >> 6) & 511;            // output column
        int m   = idx >> 15;                   // which matrix
        const float* W = (m == 0) ? Wq : (m == 1) ? Wk : Wv;
        Wt[idx] = (half_t)W[(size_t)row * DALL + col];
    } else {
        int j   = idx - 3 * DALL * D_MODEL;    // 0..32767
        int k   = j & 511;
        int col = j >> 9;
        Wot[j] = (half_t)Wo[(size_t)k * D_MODEL + col];
    }
}

// ---------------------------------------------------------------------------
// Kernel 1: QKV projections [8192x64]@[64x512]+b -> f16 [B,H,S,64]
// grid (512, 3), block 128 (4 waves); wave w covers cols [w*128, w*128+128)
// ---------------------------------------------------------------------------
__global__ __launch_bounds__(128) void rpe_proj_kernel(
    const float* __restrict__ q, const float* __restrict__ k, const float* __restrict__ v,
    const half_t* __restrict__ Wt,
    const float* __restrict__ bq, const float* __restrict__ bk, const float* __restrict__ bv,
    half_t* __restrict__ qh, half_t* __restrict__ kh, half_t* __restrict__ vh) {
    const float* X; const float* bias; half_t* dst; const half_t* W;
    switch (blockIdx.y) {
        case 0:  X = q; bias = bq; dst = qh; W = Wt;                      break;
        case 1:  X = k; bias = bk; dst = kh; W = Wt + DALL * D_MODEL;     break;
        default: X = v; bias = bv; dst = vh; W = Wt + 2 * DALL * D_MODEL; break;
    }
    int tile = blockIdx.x;
    int wid = threadIdx.x >> 5, lane = threadIdx.x & 31;
    int hf = lane >> 4, n = lane & 15;

    const float* arow = X + (size_t)tile * 16 * 64;
    v16h a0 = load_a_f32v(arow, 64, lane, 0);
    v16h a1 = load_a_f32v(arow, 64, lane, 32);

    int colw = wid * 128;
#pragma unroll
    for (int nt = 0; nt < 8; ++nt) {
        int colbase = colw + nt * 16;
        const half_t* wb = W + (size_t)colbase * 64;   // column-major block
        v8f acc = {};
        acc = wmma_f16(a0, load_b16_vec(wb, 64, lane, 0), acc);
        acc = wmma_f16(a1, load_b16_vec(wb, 64, lane, 32), acc);
        int col = colbase + n;
        int hh = col >> 6, dd = col & 63;
        float bb = bias[col];
#pragma unroll
        for (int r = 0; r < 8; ++r) {
            int rowg = tile * 16 + r + 8 * hf;
            int b_ = rowg >> 10, s_ = rowg & 1023;
            dst[(((size_t)(b_ * N_HEAD + hh) * SEQ + s_) << 6) + dd] = (half_t)(acc[r] + bb);
        }
    }
}

// ---------------------------------------------------------------------------
// Kernel 2: per-row relative bias rel[b,h,s] = q1.k_r32 + k1.q_r32
// ---------------------------------------------------------------------------
__global__ void rpe_relbias_kernel(const half_t* __restrict__ qh, const half_t* __restrict__ kh,
                                   const float* __restrict__ kr32, const float* __restrict__ qr32,
                                   float* __restrict__ rel) {
    int i = blockIdx.x * 256 + threadIdx.x;  // < 65536 rows
    const u32x4* qv = (const u32x4*)(qh + (size_t)i * 64);
    const u32x4* kv = (const u32x4*)(kh + (size_t)i * 64);
    float acc = 0.f;
#pragma unroll
    for (int c = 0; c < 8; ++c) {
        union { u32x4 q; half_t h[8]; } a, b;
        a.q = qv[c]; b.q = kv[c];
#pragma unroll
        for (int e = 0; e < 8; ++e)
            acc = fmaf((float)a.h[e], kr32[c * 8 + e], fmaf((float)b.h[e], qr32[c * 8 + e], acc));
    }
    rel[i] = acc;
}

// ---------------------------------------------------------------------------
// Kernel 3: fused flash attention. grid (B*H=64, S/64=16), block 128 (4 waves).
// Async double-buffered K staging (global_load_async_to_lds_b128 / ASYNCcnt),
// V staged transposed for vector ds B-fragment reads.
// ---------------------------------------------------------------------------
__global__ __launch_bounds__(128) void rpe_flash_kernel(
    const half_t* __restrict__ qh, const half_t* __restrict__ kh, const half_t* __restrict__ vh,
    const float* __restrict__ rel, half_t* __restrict__ o_hs) {
    __shared__ half_t Kbuf[2][32 * 64];   // [key][d]   4 KB each
    __shared__ half_t Vt[2][64 * 32];     // [d][key]   4 KB each (transposed)
    __shared__ half_t Pt[4][16 * 32];     // per-wave P tile

    int bh = blockIdx.x, qt = blockIdx.y;
    int tid = threadIdx.x;
    int wid = tid >> 5, lane = tid & 31;
    int hf = lane >> 4, n = lane & 15;
    half_t* Pw = &Pt[wid][0];

    const half_t* kbase_g = kh + ((size_t)bh * SEQ) * 64;
    const half_t* vbase_g = vh + ((size_t)bh * SEQ) * 64;

    int row0 = qt * 64 + wid * 16;
    const half_t* qrow = qh + ((size_t)bh * SEQ + row0) * 64;
    v16h qa0 = load_a16_vec(qrow, 64, lane, 0);
    v16h qa1 = load_a16_vec(qrow, 64, lane, 32);

    float relc[8];
#pragma unroll
    for (int r = 0; r < 8; ++r)
        relc[r] = rel[(size_t)bh * SEQ + row0 + r + 8 * hf] * INVSCALE;

    float mrow[8], lrow[8];
    v8f oacc[4];
#pragma unroll
    for (int r = 0; r < 8; ++r) { mrow[r] = -1e30f; lrow[r] = 0.f; }
#pragma unroll
    for (int nt = 0; nt < 4; ++nt) oacc[nt] = (v8f){};

    // ---- staging helpers -------------------------------------------------
    auto stageK = [&](int kt, int bufi) {
        uint32_t lo = lds_off(&Kbuf[bufi][0]) + (uint32_t)tid * 32;
        uint64_t ga = (uint64_t)(uintptr_t)(kbase_g + (size_t)kt * 32 * 64) + (uint64_t)tid * 32;
        asm volatile(
            "global_load_async_to_lds_b128 %0, %1, off\n\t"
            "global_load_async_to_lds_b128 %0, %1, off offset:16"
            :: "v"(lo), "v"(ga) : "memory");
    };
    auto stageV = [&](int kt, int bufi) {
        int key = tid >> 2;            // 0..31
        int dg  = (tid & 3) * 16;      // 0,16,32,48
        const half_t* src = vbase_g + ((size_t)(kt * 32 + key)) * 64 + dg;
        union { u32x4 q; half_t h[8]; } c0, c1;
        c0.q = *(const u32x4*)(src);
        c1.q = *(const u32x4*)(src + 8);
        half_t* dstv = &Vt[bufi][0];
#pragma unroll
        for (int e = 0; e < 8; ++e) {
            dstv[(dg + e) * 32 + key]     = c0.h[e];
            dstv[(dg + 8 + e) * 32 + key] = c1.h[e];
        }
    };

    stageK(0, 0);
    stageV(0, 0);

    for (int kt = 0; kt < SEQ / 32; ++kt) {
        int cur = kt & 1;
        asm volatile("s_wait_asynccnt 0x0" ::: "memory");
        __syncthreads();                       // K[cur] + V[cur] visible to all waves
        if (kt + 1 < SEQ / 32) {               // overlap next tile with compute
            stageK(kt + 1, cur ^ 1);
            stageV(kt + 1, cur ^ 1);
        }

        // scores: Q[16x64] @ K^T -> two 16x16 key tiles (K operands from LDS, Q from registers)
        const half_t* kb0 = &Kbuf[cur][0];
        v8f s0 = {}, s1 = {};
        s0 = wmma_f16(qa0, load_b16_vec(kb0, 64, lane, 0), s0);
        s0 = wmma_f16(qa1, load_b16_vec(kb0, 64, lane, 32), s0);
        const half_t* kb1 = kb0 + 16 * 64;
        s1 = wmma_f16(qa0, load_b16_vec(kb1, 64, lane, 0), s1);
        s1 = wmma_f16(qa1, load_b16_vec(kb1, 64, lane, 32), s1);

        float facs[8];
#pragma unroll
        for (int r = 0; r < 8; ++r) {
            float v0 = s0[r] * INVSCALE + relc[r];
            float v1 = s1[r] * INVSCALE + relc[r];
            float mx = fmaxf(mrow[r], hmax16(fmaxf(v0, v1)));
            float fac = __expf(mrow[r] - mx);
            float p0 = __expf(v0 - mx);
            float p1 = __expf(v1 - mx);
            lrow[r] = lrow[r] * fac + hsum16(p0 + p1);
            mrow[r] = mx;
            facs[r] = fac;
            Pw[(r + 8 * hf) * 32 + n]      = (half_t)p0;
            Pw[(r + 8 * hf) * 32 + 16 + n] = (half_t)p1;
        }
#pragma unroll
        for (int nt = 0; nt < 4; ++nt)
#pragma unroll
            for (int r = 0; r < 8; ++r) oacc[nt][r] *= facs[r];

        __syncthreads();                       // publish P (cross-lane via LDS)

        v16h pa = load_a16_vec(Pw, 32, lane, 0);
        const half_t* vt = &Vt[cur][0];
#pragma unroll
        for (int nt = 0; nt < 4; ++nt)
            oacc[nt] = wmma_f16(pa, load_b16_vec(vt + nt * 16 * 32, 32, lane, 0), oacc[nt]);
    }

    // normalize, write f16 into [B*S, 512] (col = h*64 + d)
    int b_ = bh >> 3, h_ = bh & 7;
#pragma unroll
    for (int r = 0; r < 8; ++r) {
        float inv_l = 1.f / lrow[r];
        int s_ = row0 + r + 8 * hf;
#pragma unroll
        for (int nt = 0; nt < 4; ++nt)
            o_hs[((size_t)(b_ * SEQ + s_)) * DALL + h_ * 64 + nt * 16 + n] =
                (half_t)(oacc[nt][r] * inv_l);
    }
}

// ---------------------------------------------------------------------------
// Kernel 4: output projection [8192x512]@[512x64]+bo -> fp32 out
// grid 512, block 128 (4 waves); wave w = n-tile w
// ---------------------------------------------------------------------------
__global__ __launch_bounds__(128) void rpe_outproj_kernel(
    const half_t* __restrict__ Xo, const half_t* __restrict__ Wot,
    const float* __restrict__ bo, float* __restrict__ out) {
    int tile = blockIdx.x;
    int wid = threadIdx.x >> 5, lane = threadIdx.x & 31;
    int hf = lane >> 4, n = lane & 15;
    int colbase = wid * 16;

    const half_t* arow = Xo + (size_t)tile * 16 * DALL;
    const half_t* wb = Wot + (size_t)colbase * DALL;   // column-major block
    v8f acc = {};
#pragma unroll
    for (int ks = 0; ks < 16; ++ks) {
        int kb = ks * 32;
        acc = wmma_f16(load_a16_vec(arow, DALL, lane, kb),
                       load_b16_vec(wb, DALL, lane, kb), acc);
    }
    int col = colbase + n;
    float bb = bo[col];
#pragma unroll
    for (int r = 0; r < 8; ++r)
        out[((size_t)(tile * 16 + r + 8 * hf)) * D_MODEL + col] = acc[r] + bb;
}

// ---------------------------------------------------------------------------
extern "C" void kernel_launch(void* const* d_in, const int* in_sizes, int n_in,
                              void* d_out, int out_size, void* d_ws, size_t ws_size,
                              hipStream_t stream) {
    (void)in_sizes; (void)n_in; (void)out_size; (void)ws_size;
    const float* query = (const float*)d_in[0];
    const float* key_t = (const float*)d_in[1];
    const float* value = (const float*)d_in[2];
    const float* Wq  = (const float*)d_in[3];
    const float* bq  = (const float*)d_in[4];
    const float* Wk  = (const float*)d_in[5];
    const float* bk  = (const float*)d_in[6];
    const float* Wv  = (const float*)d_in[7];
    const float* bv  = (const float*)d_in[8];
    const float* Wo  = (const float*)d_in[9];
    const float* bo  = (const float*)d_in[10];
    const float* Wkr = (const float*)d_in[11];
    const float* bkr = (const float*)d_in[12];
    const float* Wqr = (const float*)d_in[13];
    const float* bqr = (const float*)d_in[14];
    const float* p   = (const float*)d_in[15];
    float* out = (float*)d_out;

    // workspace carve-up (all f16 tensors 16B aligned)
    const size_t HT = (size_t)BATCH * N_HEAD * SEQ * 64;     // 4M elems
    char* ws = (char*)d_ws;
    half_t* qh  = (half_t*)ws;
    half_t* kh  = qh + HT;
    half_t* vh  = kh + HT;
    half_t* ohs = vh + HT;                                   // [8192 x 512] f16
    half_t* Wt  = ohs + (size_t)ROWS * DALL;                 // [3][512][64] f16
    half_t* Wot = Wt + 3 * DALL * D_MODEL;                   // [64][512] f16
    float*  rel = (float*)(Wot + D_MODEL * DALL);            // [65536] f32
    float*  kr32 = rel + (size_t)BATCH * N_HEAD * SEQ;
    float*  qr32 = kr32 + 64;

    rpe_relvec_kernel<<<1, 64, 0, stream>>>(p, Wkr, bkr, Wqr, bqr, kr32, qr32);

    rpe_wtrans_kernel<<<(3 * DALL * D_MODEL + D_MODEL * DALL) / 256, 256, 0, stream>>>(
        Wq, Wk, Wv, Wo, Wt, Wot);

    rpe_proj_kernel<<<dim3(ROWS / 16, 3), 128, 0, stream>>>(
        query, key_t, value, Wt, bq, bk, bv, qh, kh, vh);

    rpe_relbias_kernel<<<(BATCH * N_HEAD * SEQ) / 256, 256, 0, stream>>>(
        qh, kh, kr32, qr32, rel);

    rpe_flash_kernel<<<dim3(BATCH * N_HEAD, SEQ / 64), 128, 0, stream>>>(
        qh, kh, vh, rel, ohs);

    rpe_outproj_kernel<<<ROWS / 16, 128, 0, stream>>>(ohs, Wot, bo, out);
}
